// MGNBlock_72361609003663
// MI455X (gfx1250) — compile-verified, matrix operand order checked by reading, not run on hardware
//
#include <hip/hip_runtime.h>

// ---------------------------------------------------------------------------
// MGN block: fv_agg = fv @ adj ; out = W @ [fv; fv_agg] ; BN(train) ; ReLU
// bf16 WMMA (v_wmma_f32_16x16x32_bf16) with f32 accumulation, fully fused.
// LDS position-major + XOR swizzle: B fragments = 2x ds_load_b128,
// staging + agg-D writes = ds_store_b128, packed cvt via inline asm.
// ---------------------------------------------------------------------------

typedef __attribute__((ext_vector_type(16))) __bf16 v16bf;
typedef __attribute__((ext_vector_type(8)))  float  v8f;

union BfVec {
    v16bf v;
    __bf16 h[16];
    unsigned short s[16];
    unsigned u[8];
    uint4 q[2];
};

// One v_cvt_pk_bf16_f32 per bf16 pair (compiler refuses to fuse scalar casts).
__device__ __forceinline__ unsigned pk2(float a, float b) {
    unsigned r;
    asm("v_cvt_pk_bf16_f32 %0, %1, %2" : "=v"(r) : "v"(a), "v"(b));
    return r;
}

// ISA 7.12.2: element e (0..15) of a 16-bit A/B lane payload maps to
// K = 8*hi + e            (e < 8)
// K = 16 + 8*hi + (e-8)   (e >= 8)
__device__ __forceinline__ int kmap(int e, int hi) {
    int vg = e >> 1;
    return ((vg & 3) << 1) + ((vg >> 2) << 4) + (hi << 3) + (e & 1);
}

// XOR swizzle on aligned 8-channel groups: preserves contiguity/alignment of
// every 8-wide fragment access while decorrelating LDS banks across rows.
__device__ __forceinline__ int swz(int p, int c) {
    return c ^ ((p & 7) << 3);
}

#define NB   32
#define CIN  64
#define TT   288
#define VV   25
#define CO   64
#define K2   128
#define TB   8              // t-rows per workgroup
#define PP   (TB * 32)      // 256 padded positions per workgroup (V 25->32)
#define NTIL (PP / 16)      // 16 N-tiles

__global__ void __launch_bounds__(128) mgn_init_stats(float* stats) {
    int i = threadIdx.x;
    stats[i] = 0.0f;        // [0..63] sum, [64..127] sumsq
    stats[128 + i] = 0.0f;
}

__global__ void __launch_bounds__(256) mgn_pass1(
    const float* __restrict__ fv, const float* __restrict__ adj,
    const float* __restrict__ W, float* __restrict__ out,
    float* __restrict__ stats)
{
    // position-major, XOR-swizzled: [p][c], 32 KB each
    __shared__ __align__(16) unsigned short Xs[PP][CIN];  // fv tile (bf16)
    __shared__ __align__(16) unsigned short As[PP][CIN];  // agg tile (bf16)

    const int n   = blockIdx.x / (TT / TB);
    const int t0  = (blockIdx.x % (TT / TB)) * TB;
    const int tid = threadIdx.x;
    const int wave = tid >> 5;
    const int lane = tid & 31;
    const int hi = lane >> 4;
    const int ln = lane & 15;

    // ---- stage fv tile (fp32 -> bf16), zero V-pad -------------------------
    // thread owns position p = tid; packs 8 channels -> one ds_store_b128.
    // Channel offsets are int constants -> immediate-offset global loads.
    {
        const int p = tid;
        const int t = p >> 5, v = p & 31;
        const bool valid = (v < VV);
        const int vc = valid ? v : 0;              // clamp: no OOB reads
        const unsigned msk = valid ? 0xFFFFFFFFu : 0u;
        const float* src = fv + (((size_t)n * CIN) * TT + (t0 + t)) * VV + vc;
        #pragma unroll
        for (int g = 0; g < 8; ++g) {
            const int cb = g * 8;
            uint4 pk;
            pk.x = pk2(src[(cb + 0) * (TT * VV)], src[(cb + 1) * (TT * VV)]) & msk;
            pk.y = pk2(src[(cb + 2) * (TT * VV)], src[(cb + 3) * (TT * VV)]) & msk;
            pk.z = pk2(src[(cb + 4) * (TT * VV)], src[(cb + 5) * (TT * VV)]) & msk;
            pk.w = pk2(src[(cb + 6) * (TT * VV)], src[(cb + 7) * (TT * VV)]) & msk;
            *reinterpret_cast<uint4*>(&Xs[p][swz(p, cb)]) = pk;
        }
    }
    __syncthreads();

    // ---- phase A: Agg = X @ adj  (per t-row, wave w handles t = w) --------
    {
        const int t = wave;
        v16bf Badj[2];
        #pragma unroll
        for (int nt = 0; nt < 2; ++nt) {
            BfVec bb;
            int y = nt * 16 + ln;
            #pragma unroll
            for (int j = 0; j < 8; ++j) {
                int v0 = kmap(2 * j, hi);               // pair (v0, v0+1)
                float a0 = (v0     < VV && y < VV) ? adj[v0 * VV + y]       : 0.0f;
                float a1 = (v0 + 1 < VV && y < VV) ? adj[(v0 + 1) * VV + y] : 0.0f;
                bb.u[j] = pk2(a0, a1);
            }
            Badj[nt] = bb.v;
        }
        #pragma unroll
        for (int mt = 0; mt < 4; ++mt) {
            BfVec aa;                                   // transpose read: u16s
            int c = mt * 16 + ln;                       // M index = channel
            #pragma unroll
            for (int e = 0; e < 16; ++e) {
                int p = t * 32 + kmap(e, hi);           // pad already zero
                aa.s[e] = Xs[p][swz(p, c)];
            }
            #pragma unroll
            for (int nt = 0; nt < 2; ++nt) {
                v8f acc = {};
                acc = __builtin_amdgcn_wmma_f32_16x16x32_bf16(
                        false, aa.v, false, Badj[nt], (short)0, acc, false, false);
                // D fragment = 8 consecutive channels per lane -> one b128 store
                int p = t * 32 + nt * 16 + ln;
                int c0 = mt * 16 + 8 * hi;
                uint4 pk;
                pk.x = pk2(acc[0], acc[1]);
                pk.y = pk2(acc[2], acc[3]);
                pk.z = pk2(acc[4], acc[5]);
                pk.w = pk2(acc[6], acc[7]);
                *reinterpret_cast<uint4*>(&As[p][swz(p, c0)]) = pk;
            }
        }
    }
    __syncthreads();

    // ---- phase B: out = W @ [X ; Agg]  (K = 128, 4 k-steps of 32) ---------
    {
        const int mt = wave & 3;           // M-tile (16 output channels)
        const int nh = wave >> 2;          // which half of the 16 N-tiles
        const int o  = mt * 16 + ln;       // A row = output channel

        // Each lane's K-run = two contiguous 8-float spans -> 4x b128 loads
        v16bf Aw[4];
        #pragma unroll
        for (int ks = 0; ks < 4; ++ks) {
            const float* wrow = W + o * K2 + ks * 32 + 8 * hi;
            float4 a0 = *reinterpret_cast<const float4*>(wrow);
            float4 a1 = *reinterpret_cast<const float4*>(wrow + 4);
            float4 b0 = *reinterpret_cast<const float4*>(wrow + 16);
            float4 b1 = *reinterpret_cast<const float4*>(wrow + 20);
            BfVec aa;
            aa.u[0] = pk2(a0.x, a0.y); aa.u[1] = pk2(a0.z, a0.w);
            aa.u[2] = pk2(a1.x, a1.y); aa.u[3] = pk2(a1.z, a1.w);
            aa.u[4] = pk2(b0.x, b0.y); aa.u[5] = pk2(b0.z, b0.w);
            aa.u[6] = pk2(b1.x, b1.y); aa.u[7] = pk2(b1.z, b1.w);
            Aw[ks] = aa.v;
        }

        float csum[8], csq[8];
        #pragma unroll
        for (int r = 0; r < 8; ++r) { csum[r] = 0.0f; csq[r] = 0.0f; }

        #pragma unroll
        for (int nti = 0; nti < NTIL / 2; ++nti) {
            const int pbase = (nh * (NTIL / 2) + nti) * 16;
            const int p = pbase + ln;
            v8f acc = {};
            #pragma unroll
            for (int ks = 0; ks < 4; ++ks) {
                const unsigned short (*buf)[CIN] = (ks < 2) ? Xs : As;
                const int cb = (ks & 1) * 32;
                // two contiguous 8xbf16 runs per lane -> 2x ds_load_b128
                BfVec bb;
                bb.q[0] = *reinterpret_cast<const uint4*>(
                              &buf[p][swz(p, cb + 8 * hi)]);
                bb.q[1] = *reinterpret_cast<const uint4*>(
                              &buf[p][swz(p, cb + 16 + 8 * hi)]);
                acc = __builtin_amdgcn_wmma_f32_16x16x32_bf16(
                        false, Aw[ks], false, bb.v, (short)0, acc, false, false);
            }
            const int t = p >> 5, v = p & 31;
            if (v < VV) {
                // 8 channel stores share a base; stride 7200 -> imm offsets
                float* obase = out +
                    (((size_t)n * CO + (mt * 16 + 8 * hi)) * TT + (t0 + t)) * VV + v;
                #pragma unroll
                for (int r = 0; r < 8; ++r) {
                    float val = acc[r];
                    obase[r * (TT * VV)] = val;
                    csum[r] += val;
                    csq[r]  += val * val;
                }
            }
        }

        // reduce across the 16 lanes sharing each channel, then atomics
        #pragma unroll
        for (int m = 1; m < 16; m <<= 1) {
            #pragma unroll
            for (int r = 0; r < 8; ++r) {
                csum[r] += __shfl_xor(csum[r], m, 32);
                csq[r]  += __shfl_xor(csq[r],  m, 32);
            }
        }
        if (ln == 0) {
            #pragma unroll
            for (int r = 0; r < 8; ++r) {
                int oo = mt * 16 + r + 8 * hi;
                atomicAdd(&stats[oo],      csum[r]);
                atomicAdd(&stats[64 + oo], csq[r]);
            }
        }
    }
}

__global__ void __launch_bounds__(64) mgn_pass2(
    float* __restrict__ stats, const float* __restrict__ gamma,
    const float* __restrict__ beta)
{
    int c = threadIdx.x;
    const float cnt = (float)NB * TT * VV;
    float mean = stats[c] / cnt;
    float var  = stats[64 + c] / cnt - mean * mean;
    float sc   = gamma[c] * rsqrtf(var + 1e-5f);
    stats[128 + c] = sc;
    stats[192 + c] = beta[c] - mean * sc;
}

__global__ void __launch_bounds__(256) mgn_pass3(
    float* __restrict__ out, const float* __restrict__ stats)
{
    size_t i = (size_t)blockIdx.x * 256 + threadIdx.x;   // over float4 units
    int ch = (int)((i / (7200 / 4)) & (CO - 1));          // 7200 % 4 == 0
    float sc = stats[128 + ch], sh = stats[192 + ch];
    float4* o4 = reinterpret_cast<float4*>(out);
    float4 v = o4[i];
    v.x = fmaxf(fmaf(v.x, sc, sh), 0.0f);
    v.y = fmaxf(fmaf(v.y, sc, sh), 0.0f);
    v.z = fmaxf(fmaf(v.z, sc, sh), 0.0f);
    v.w = fmaxf(fmaf(v.w, sc, sh), 0.0f);
    o4[i] = v;
}

extern "C" void kernel_launch(void* const* d_in, const int* in_sizes, int n_in,
                              void* d_out, int out_size, void* d_ws, size_t ws_size,
                              hipStream_t stream) {
    const float* fv    = (const float*)d_in[0];
    const float* adj   = (const float*)d_in[1];
    const float* W     = (const float*)d_in[2];
    // d_in[3] = bias: exactly cancelled by training-mode BN mean subtraction.
    const float* gamma = (const float*)d_in[4];
    const float* beta  = (const float*)d_in[5];
    float* out   = (float*)d_out;
    float* stats = (float*)d_ws;     // 256 floats: sum, sumsq, scale, shift

    mgn_init_stats<<<dim3(1), dim3(128), 0, stream>>>(stats);
    mgn_pass1<<<dim3(NB * (TT / TB)), dim3(256), 0, stream>>>(fv, adj, W, out, stats);
    mgn_pass2<<<dim3(1), dim3(64), 0, stream>>>(stats, gamma, beta);
    mgn_pass3<<<dim3((NB * CO * TT * VV) / 4 / 256), dim3(256), 0, stream>>>(out, stats);
}
